// SemiMarkovModule_29162827940490
// MI455X (gfx1250) — matrix operand order repaired and checked by reading, not compiled, not run on hardware
//
#include <hip/hip_runtime.h>
#include <hip/hip_bf16.h>

// ---------------- problem constants (match setup_inputs) ----------------
#define BB   64      // batch
#define TT   2048    // time steps
#define CC   64      // states / components
#define DD   128     // feature dim
#define KK   20      // max span length
#define BT   (BB*TT) // 131072 rows

// ---------------- workspace layout (bytes) ----------------
#define OFF_W16    ((size_t)0)                    // 128*128 f16 = 32 KB
#define OFF_NU16   ((size_t)32768)                // 64*128  f16 = 16 KB
#define OFF_QMU    ((size_t)49152)                // 64 f32
#define OFF_CONST  ((size_t)49664)                // scalars
#define OFF_CHOL   ((size_t)50176)                // 128*128 f32 = 64 KB
#define OFF_WF     ((size_t)115712)               // 128*128 f32 = 64 KB
#define OFF_NUF    ((size_t)181248)               // 64*128  f32 = 32 KB
#define OFF_Z16    ((size_t)262144)               // BT*128 f16 = 32 MiB
#define OFF_EM     ((size_t)(262144 + (size_t)BT*DD*2)) // BT*64 f32 = 32 MiB

typedef __attribute__((ext_vector_type(16))) _Float16 v16h;
typedef __attribute__((ext_vector_type(8)))  _Float16 v8h;
typedef __attribute__((ext_vector_type(8)))  float    v8f;

// ---------------- WMMA fragment loaders (gfx1250 wave32 layouts) --------
// A (16x32 f16): lane -> row (lane&15); K = {b..b+7} U {b+16..b+23}, b=8*(lane>>4)
__device__ inline v16h load_fragA_f16(const _Float16* __restrict__ base, int ld, int lane) {
  const int row = lane & 15;
  const int k0  = (lane >> 4) << 3;           // 0 or 8
  const _Float16* p = base + (size_t)row * ld + k0;
  v8h lo = *(const v8h*)(p);
  v8h hi = *(const v8h*)(p + 16);
  v16h r;
#pragma unroll
  for (int i = 0; i < 8; ++i) { r[i] = lo[i]; r[8 + i] = hi[i]; }
  return r;
}

__device__ inline v16h load_fragA_f32(const float* __restrict__ base, int ld, int lane) {
  const int row = lane & 15;
  const int k0  = (lane >> 4) << 3;
  const float* p = base + (size_t)row * ld + k0;
  float4 a0 = *(const float4*)(p);
  float4 a1 = *(const float4*)(p + 4);
  float4 b0 = *(const float4*)(p + 16);
  float4 b1 = *(const float4*)(p + 20);
  v16h r;
  r[0]=(_Float16)a0.x; r[1]=(_Float16)a0.y; r[2]=(_Float16)a0.z; r[3]=(_Float16)a0.w;
  r[4]=(_Float16)a1.x; r[5]=(_Float16)a1.y; r[6]=(_Float16)a1.z; r[7]=(_Float16)a1.w;
  r[8]=(_Float16)b0.x; r[9]=(_Float16)b0.y; r[10]=(_Float16)b0.z; r[11]=(_Float16)b0.w;
  r[12]=(_Float16)b1.x; r[13]=(_Float16)b1.y; r[14]=(_Float16)b1.z; r[15]=(_Float16)b1.w;
  return r;
}

// B (32x16 f16), B[k][n] = src row n (output col), K = {b..b+7} U {b+16..b+23}, b=8*(lane>>4)
__device__ inline v16h load_fragB_f16(const _Float16* __restrict__ base, int ld, int lane) {
  const int col = lane & 15;
  const int k0  = (lane >> 4) << 3;           // 0 or 8
  const _Float16* p = base + (size_t)col * ld + k0;
  v8h lo = *(const v8h*)(p);
  v8h hi = *(const v8h*)(p + 16);
  v16h r;
#pragma unroll
  for (int i = 0; i < 8; ++i) { r[i] = lo[i]; r[8 + i] = hi[i]; }
  return r;
}

// ---------------- K0: Cholesky, W = inv(L), Nu, q_mu, const ----------------
__global__ void prep_kernel(const float* __restrict__ cov,
                            const float* __restrict__ means,
                            char* __restrict__ ws) {
  float*     A    = (float*)(ws + OFF_CHOL);
  float*     Wf   = (float*)(ws + OFF_WF);
  float*     Nuf  = (float*)(ws + OFF_NUF);
  _Float16*  W16  = (_Float16*)(ws + OFF_W16);
  _Float16*  Nu16 = (_Float16*)(ws + OFF_NU16);
  float*     qmu  = (float*)(ws + OFF_QMU);
  float*     cst  = (float*)(ws + OFF_CONST);
  const int tid = threadIdx.x;   // 0..127

  for (int i = tid; i < DD*DD; i += 128) A[i] = cov[i];
  __syncthreads();

  // in-place Cholesky (lower)
  for (int j = 0; j < DD; ++j) {
    if (tid == 0) {
      float s = A[j*DD + j];
      for (int k = 0; k < j; ++k) s -= A[j*DD + k] * A[j*DD + k];
      A[j*DD + j] = sqrtf(fmaxf(s, 1e-30f));
    }
    __syncthreads();
    const float dinv = 1.0f / A[j*DD + j];
    if (tid > j) {
      const int i = tid;
      float s = A[i*DD + j];
      for (int k = 0; k < j; ++k) s -= A[i*DD + k] * A[j*DD + k];
      A[i*DD + j] = s * dinv;
    }
    __syncthreads();
  }

  // W = inv(L) by forward substitution; thread owns one column
  {
    const int ccol = tid;
    for (int i = 0; i < DD; ++i) {
      if (i < ccol) { Wf[i*DD + ccol] = 0.0f; continue; }
      float s = (i == ccol) ? 1.0f : 0.0f;
      for (int k = ccol; k < i; ++k) s -= A[i*DD + k] * Wf[k*DD + ccol];
      Wf[i*DD + ccol] = s / A[i*DD + i];
    }
  }
  __syncthreads();

  for (int i = tid; i < DD*DD; i += 128) W16[i] = (_Float16)Wf[i];
  if (tid == 0) {
    float ldh = 0.0f;
    for (int i = 0; i < DD; ++i) ldh += __logf(A[i*DD + i]);
    cst[0] = (float)DD * 1.8378770664093453f /* log(2*pi) */ + 2.0f * ldh;
  }
  __syncthreads();

  // Nu[c][j] = sum_d means[c][d] * W[j][d]
  for (int o = tid; o < CC*DD; o += 128) {
    const int c2 = o >> 7, j = o & (DD - 1);
    float s = 0.0f;
    for (int dd = 0; dd < DD; ++dd) s += means[c2*DD + dd] * Wf[j*DD + dd];
    Nuf[o] = s; Nu16[o] = (_Float16)s;
  }
  __syncthreads();
  if (tid < CC) {
    float s = 0.0f;
    for (int j = 0; j < DD; ++j) { float v = Nuf[tid*DD + j]; s += v * v; }
    qmu[tid] = s;
  }
}

// ---------------- K1: Z = X @ W^T  (f16 WMMA, f32 accum, f16 out) ----------
__global__ void __launch_bounds__(256)
zgemm_kernel(const float* __restrict__ X, const char* __restrict__ ws) {
  const _Float16* W16 = (const _Float16*)(ws + OFF_W16);
  _Float16*       Z16 = (_Float16*)(ws + OFF_Z16);
  const int wid  = blockIdx.x * 8 + (threadIdx.x >> 5); // 16384 waves
  const int lane = threadIdx.x & 31;
  const int mtile = wid >> 1;                           // 8192 M-tiles
  const int jb    = wid & 1;                            // which 64-col half of W
  const float* Abase = X + (size_t)mtile * 16 * DD;

  v8f acc[4];
#pragma unroll
  for (int nt = 0; nt < 4; ++nt)
#pragma unroll
    for (int i = 0; i < 8; ++i) acc[nt][i] = 0.0f;

#pragma unroll
  for (int kc = 0; kc < 4; ++kc) {
    v16h a = load_fragA_f32(Abase + kc * 32, DD, lane);
#pragma unroll
    for (int nt = 0; nt < 4; ++nt) {
      const _Float16* Bb = W16 + (size_t)(jb * 64 + nt * 16) * DD + kc * 32;
      v16h b = load_fragB_f16(Bb, DD, lane);
      acc[nt] = __builtin_amdgcn_wmma_f32_16x16x32_f16(
          false, a, false, b, (short)0, acc[nt], false, false);
    }
  }
  const int colbase = jb * 64;
#pragma unroll
  for (int nt = 0; nt < 4; ++nt)
#pragma unroll
    for (int v = 0; v < 8; ++v) {
      const int row = mtile * 16 + v + ((lane >> 4) << 3);
      const int col = colbase + nt * 16 + (lane & 15);
      Z16[(size_t)row * DD + col] = (_Float16)acc[nt][v];
    }
}

// ---------------- K2: em = Z @ Nu^T - 0.5*(|z|^2 + |nu|^2 + const) --------
__global__ void __launch_bounds__(256)
emis_kernel(const char* __restrict__ ws_in, char* __restrict__ ws) {
  const _Float16* Z16  = (const _Float16*)(ws_in + OFF_Z16);
  const _Float16* Nu16 = (const _Float16*)(ws_in + OFF_NU16);
  const float*    qmu  = (const float*)(ws_in + OFF_QMU);
  const float*    cst  = (const float*)(ws_in + OFF_CONST);
  float*          em   = (float*)(ws + OFF_EM);

  const int mtile = blockIdx.x * 8 + (threadIdx.x >> 5); // 8192 tiles
  const int lane  = threadIdx.x & 31;
  const _Float16* Abase = Z16 + (size_t)mtile * 16 * DD;

  v8f acc[4];
#pragma unroll
  for (int nt = 0; nt < 4; ++nt)
#pragma unroll
    for (int i = 0; i < 8; ++i) acc[nt][i] = 0.0f;

  float qx = 0.0f;  // row |z|^2, built from the A fragments this lane loads
#pragma unroll
  for (int kc = 0; kc < 4; ++kc) {
    v16h a = load_fragA_f16(Abase + kc * 32, DD, lane);
#pragma unroll
    for (int i = 0; i < 16; ++i) { float f = (float)a[i]; qx += f * f; }
#pragma unroll
    for (int nt = 0; nt < 4; ++nt) {
      v16h b = load_fragB_f16(Nu16 + (size_t)(nt * 16) * DD + kc * 32, DD, lane);
      acc[nt] = __builtin_amdgcn_wmma_f32_16x16x32_f16(
          false, a, false, b, (short)0, acc[nt], false, false);
    }
  }
  // lane l and l+16 hold the same row -> one xor completes the row norm
  qx += __shfl_xor(qx, 16, 32);      // now lane l holds q_x of row (l&15)
  const float ct = cst[0];
#pragma unroll
  for (int nt = 0; nt < 4; ++nt) {
    const int col = nt * 16 + (lane & 15);
    const float qm = qmu[col];
#pragma unroll
    for (int v = 0; v < 8; ++v) {
      const int r   = v + ((lane >> 4) << 3);
      const float qr = __shfl(qx, r, 32);
      const int row = mtile * 16 + r;
      em[(size_t)row * CC + col] = acc[nt][v] - 0.5f * (qr + qm + ct);
    }
  }
}

// ---------------- K3: semi-Markov forward recursion (1 block / batch) ------
__global__ void __launch_bounds__(64)
hsmm_forward(const char* __restrict__ ws,
             const float* __restrict__ plr,      // poisson_log_rates (C)
             const float* __restrict__ tl,       // transition_logits (C,C)
             const float* __restrict__ il,       // init_logits (C)
             const int*   __restrict__ lengths,  // (b)
             float*       __restrict__ out) {    // (b)
  const float* em = (const float*)(ws + OFF_EM);
  __shared__ float Texp[CC * 65];    // padded stride 65: conflict-free rows
  __shared__ float lenl[KK * CC];
  __shared__ float ubuf[KK * CC];
  __shared__ float ebuf[KK * CC];
  __shared__ float colL[CC];
  __shared__ float aexp[CC];
  __shared__ float alds[CC];
  __shared__ float wred[2];

  const int b = blockIdx.x;
  const int c = threadIdx.x;         // state index, 0..63

  // init_log = log_softmax(init_logits)
  alds[c] = il[c];
  __syncthreads();
  float m = -1e30f, s = 0.0f;
  for (int i = 0; i < CC; ++i) {
    float v = alds[i];
    if (v > m) { s = s * __expf(m - v) + 1.0f; m = v; } else s += __expf(v - m);
  }
  const float initlog = alds[c] - (m + __logf(s));
  __syncthreads();

  // column LSE of masked transition logits (column = c as "from")
  m = -1e30f; s = 0.0f;
  for (int ct2 = 0; ct2 < CC; ++ct2) {
    float v = tl[ct2 * CC + c] + ((ct2 == c) ? -1e9f : 0.0f);
    if (v > m) { s = s * __expf(m - v) + 1.0f; m = v; } else s += __expf(v - m);
  }
  colL[c] = m + __logf(s);
  __syncthreads();
  for (int cf = 0; cf < CC; ++cf) {
    float v = tl[c * CC + cf] + ((c == cf) ? -1e9f : 0.0f);
    Texp[c * 65 + cf] = __expf(v - colL[cf]);
  }
  // Poisson span-length table: k*lr - exp(lr) - log(k!)
  {
    const float lr = plr[c], er = __expf(lr);
    float lf = 0.0f;
    for (int k = 1; k <= KK; ++k) {
      lf += __logf((float)k);
      lenl[(k - 1) * CC + c] = (float)k * lr - er - lf;
    }
  }
  for (int i = 0; i < KK; ++i) {
    ubuf[i * CC + c] = (i == 0) ? initlog : -1e9f;
    ebuf[i * CC + c] = 0.0f;
  }
  const int lenm1 = lengths[b] - 1;
  const float* emb = em + (size_t)b * TT * CC + c;
  float ecum = 0.0f;
  int head = 0;

  float pf[8];                        // 8-deep register prefetch pipeline
#pragma unroll
  for (int j = 0; j < 8; ++j) pf[j] = emb[(size_t)j * CC];
  __syncthreads();

  for (int tb = 0; tb < TT; tb += 8) {
    if (((tb & 127) == 0) && (tb + 136 < TT))
      __builtin_prefetch(emb + (size_t)(tb + 136) * CC, 0, 1);
#pragma unroll
    for (int j = 0; j < 8; ++j) {
      const int t = tb + j;
      const float e = pf[j];
      const int tn = t + 8;
      pf[j] = (tn < TT) ? emb[(size_t)tn * CC] : 0.0f;
      ecum += e;

      // alpha_t[c] = LSE_i( u[t-1-i] + (ecum_t - ecum_{t-1-i}) + len_log[i] )
      float am = -1e30f, as = 0.0f;
#pragma unroll
      for (int i = 0; i < KK; ++i) {
        int slot = head + i; if (slot >= KK) slot -= KK;
        const float sv = ubuf[slot * CC + c] + (ecum - ebuf[slot * CC + c])
                       + lenl[i * CC + c];
        if (sv > am) { as = as * __expf(am - sv) + 1.0f; am = sv; }
        else as += __expf(sv - am);
      }
      const float alpha = am + __logf(as);

      // global max over states (wave shfl + 2-entry LDS)
      float wm = alpha;
#pragma unroll
      for (int off = 16; off; off >>= 1) wm = fmaxf(wm, __shfl_xor(wm, off, 32));
      if ((c & 31) == 0) wred[c >> 5] = wm;
      __syncthreads();
      const float gm = fmaxf(wred[0], wred[1]);
      aexp[c] = __expf(alpha - gm);
      if (t == lenm1) alds[c] = alpha;
      __syncthreads();

      // u_t[c_to] = log( sum_cf Texp[c_to][cf] * exp(alpha[cf]-gm) ) + gm
      float dot = 0.0f;
#pragma unroll 4
      for (int cf = 0; cf < CC; ++cf) dot += Texp[c * 65 + cf] * aexp[cf];
      const float u = __logf(dot) + gm;

      if (t == lenm1 && c == 0) {     // logZ[b] = LSE_c alpha[len-1][c]
        float fm = -1e30f, fs = 0.0f;
        for (int i = 0; i < CC; ++i) {
          float v = alds[i];
          if (v > fm) { fs = fs * __expf(fm - v) + 1.0f; fm = v; }
          else fs += __expf(v - fm);
        }
        out[b] = fm + __logf(fs);
      }

      head = head ? head - 1 : KK - 1;
      ubuf[head * CC + c] = u;
      ebuf[head * CC + c] = ecum;
      __syncthreads();
    }
  }
}

// ---------------- launcher ----------------
extern "C" void kernel_launch(void* const* d_in, const int* in_sizes, int n_in,
                              void* d_out, int out_size, void* d_ws, size_t ws_size,
                              hipStream_t stream) {
  (void)in_sizes; (void)n_in; (void)out_size; (void)ws_size;
  const float* features = (const float*)d_in[0];   // (b,T,d) f32
  const int*   lengths  = (const int*)d_in[1];     // (b) i32
  const float* plr      = (const float*)d_in[2];   // (C) f32
  const float* means    = (const float*)d_in[3];   // (C,d) f32
  const float* cov      = (const float*)d_in[4];   // (d,d) f32
  const float* tl       = (const float*)d_in[5];   // (C,C) f32
  const float* il       = (const float*)d_in[6];   // (C) f32
  char* ws = (char*)d_ws;
  float* out = (float*)d_out;

  prep_kernel <<<1, 128, 0, stream>>>(cov, means, ws);
  zgemm_kernel<<<2048, 256, 0, stream>>>(features, ws);   // 16384 waves
  emis_kernel <<<1024, 256, 0, stream>>>(ws, ws);         //  8192 waves
  hsmm_forward<<<BB, CC, 0, stream>>>(ws, plr, tl, il, lengths, out);
}
